// TopologyBranch_4483945857302
// MI455X (gfx1250) — matrix operand order repaired and checked by reading, not compiled, not run on hardware
//
#include <hip/hip_runtime.h>

typedef __attribute__((ext_vector_type(16))) _Float16 v16h;
typedef __attribute__((ext_vector_type(8)))  _Float16 v8h;
typedef __attribute__((ext_vector_type(8)))  float    v8f;
typedef __attribute__((ext_vector_type(4)))  int      v4i;

typedef __attribute__((address_space(1))) v4i gbl_v4i;
typedef __attribute__((address_space(3))) v4i lds_v4i;

#if __has_builtin(__builtin_amdgcn_global_load_async_to_lds_b128) && \
    __has_builtin(__builtin_amdgcn_s_wait_asynccnt)
#define GFX1250_HAS_ASYNC 1
#endif

__device__ __forceinline__ v8f wmma_f16(v16h a, v16h b, v8f c) {
    // D = A(16x32) * B(32x16) + C, f32 accumulate
    return __builtin_amdgcn_wmma_f32_16x16x32_f16(
        /*neg_a=*/false, a, /*neg_b=*/false, b,
        /*c_mod=*/(short)0, c, /*reuse_a=*/false, /*reuse_b=*/false);
}

// 16-byte global -> LDS copy; async (ASYNCcnt-tracked) when available.
__device__ __forceinline__ void copy16_g2l(void* lds, const void* gl) {
#ifdef GFX1250_HAS_ASYNC
    __builtin_amdgcn_global_load_async_to_lds_b128(
        (gbl_v4i*)(gl),
        (lds_v4i*)(lds),
        /*offset=*/0, /*cpol=*/0);
#else
    *(v8h*)lds = *(const v8h*)gl;
#endif
}

template <int N>
__device__ __forceinline__ void wait_async() {
#ifdef GFX1250_HAS_ASYNC
    __builtin_amdgcn_s_wait_asynccnt(N);
#endif
}

// op: 0=min, 1=max, 2=sum. blockDim.x == 256.
__device__ __forceinline__ float blockReduce(float v, float* sbuf, int op) {
    const int tid = threadIdx.x;
    sbuf[tid] = v;
    __syncthreads();
    for (int s = 128; s > 0; s >>= 1) {
        if (tid < s) {
            float a = sbuf[tid], b = sbuf[tid + s];
            sbuf[tid] = (op == 0) ? fminf(a, b) : (op == 1) ? fmaxf(a, b) : (a + b);
        }
        __syncthreads();
    }
    float r = sbuf[0];
    __syncthreads();
    return r;
}

// ---------------------------------------------------------------------------
// Kernel 1: persistence image per (batch, homology-dim) block.
// img(64x64) = sum_p w_p * kb[p][i] * kp[p][j]  ==  A'(64xP) @ B(Px64)
// 8 waves/block, each wave owns two 16x16 output tiles (same M sub-tile, so
// the A fragment is shared); P chunked by 32 (WMMA K).
// ---------------------------------------------------------------------------
__global__ __launch_bounds__(256)
void pers_image_kernel(const float* __restrict__ pairs0,
                       const float* __restrict__ pairs1,
                       _Float16* __restrict__ feats)
{
    const int tid = threadIdx.x;
    const int b   = blockIdx.x >> 1;
    const int h   = blockIdx.x & 1;
    const int P   = h ? 512 : 1024;
    const float* pairs = h ? (pairs1 + (size_t)b * 512 * 2)
                           : (pairs0 + (size_t)b * 1024 * 2);

    __shared__ float sbuf[256];
    __shared__ __align__(64) _Float16 aLds[64][32];   // [i][p] = w * kb
    __shared__ __align__(64) _Float16 bLds[64][32];   // [j][p] = kp

    // masked min/max over valid pairs (pers > 1e-6)
    float lbmin = 1e30f, lbmax = -1e30f, lpmin = 1e30f, lpmax = -1e30f, lany = 0.f;
    for (int p = tid; p < P; p += 256) {
        float bb = pairs[2 * p + 0];
        float dd = pairs[2 * p + 1];
        float pe = dd - bb;
        if (pe > 1e-6f) {
            lany = 1.f;
            lbmin = fminf(lbmin, bb); lbmax = fmaxf(lbmax, bb);
            lpmin = fminf(lpmin, pe); lpmax = fmaxf(lpmax, pe);
        }
    }
    float bmin = blockReduce(lbmin, sbuf, 0);
    float bmax = blockReduce(lbmax, sbuf, 1);
    float pmin = blockReduce(lpmin, sbuf, 0);
    float pmax = blockReduce(lpmax, sbuf, 1);
    float any  = blockReduce(lany,  sbuf, 1);
    if (bmax - bmin < 1e-6f) { bmin = 0.f; bmax = 1.f; }
    if (pmax - pmin < 1e-6f) { pmin = 0.f; pmax = 1.f; }
    const float invDb = 1.f / (bmax - bmin + 1e-8f);
    const float invDp = 1.f / (pmax - pmin + 1e-8f);

    const int wave = tid >> 5, lane = tid & 31;
    const int lane16 = lane & 15, grp = lane >> 4;
    const int t0 = wave * 2;
    const int iT = t0 >> 2, jT0 = t0 & 3, jT1 = jT0 + 1;
    v8f acc0 = {}, acc1 = {};

    const int pp = tid & 31;        // p within chunk
    const int i0 = (tid >> 5) * 8;  // 8 grid rows per thread

    for (int c0 = 0; c0 < P; c0 += 32) {
        const int p = c0 + pp;
        float bb = pairs[2 * p + 0];
        float dd = pairs[2 * p + 1];
        float pe = dd - bb;
        float w  = (pe > 1e-6f) ? pe : 0.f;
        float bn = (bb - bmin) * invDb;
        float pn = (pe - pmin) * invDp;
        #pragma unroll
        for (int r = 0; r < 8; ++r) {
            float gi = (float)(i0 + r) * (1.0f / 63.0f);
            float db = gi - bn, dp = gi - pn;
            aLds[i0 + r][pp] = (_Float16)(w * __expf(-db * db * 0.5f));
            bLds[i0 + r][pp] = (_Float16)(__expf(-dp * dp * 0.5f));
        }
        __syncthreads();

        const _Float16* pa = &aLds[iT * 16 + lane16][0];
        v8h x = *(const v8h*)(pa + grp * 16);
        v8h y = *(const v8h*)(pa + grp * 16 + 8);
        v16h a = __builtin_shufflevector(x, y, 0,1,2,3,4,5,6,7,8,9,10,11,12,13,14,15);
        v16h bf0 = *(const v16h*)(&bLds[jT0 * 16 + lane16][grp * 16]);
        v16h bf1 = *(const v16h*)(&bLds[jT1 * 16 + lane16][grp * 16]);
        acc0 = wmma_f16(a, bf0, acc0);
        acc1 = wmma_f16(a, bf1, acc1);
        __syncthreads();
    }

    const float sc = (any > 0.f) ? 1.f : 0.f;
    _Float16* out = feats + (size_t)b * 8192 + (size_t)h * 4096;
    #pragma unroll
    for (int r = 0; r < 8; ++r) {
        int i = iT * 16 + grp * 8 + r;
        out[i * 64 + jT0 * 16 + lane16] = (_Float16)(acc0[r] * sc);
        out[i * 64 + jT1 * 16 + lane16] = (_Float16)(acc1[r] * sc);
    }
}

// ---------------------------------------------------------------------------
// Kernel 2: LDS-tiled WMMA GEMM with double-buffered async global->LDS
// staging.  C[M][N] = A[M][K](f16) * Bt[N][K](f16)^T + bias.
// Block = 8 waves = 64x64 C tile; per K-chunk(32) the block stages a 64x32 A
// tile and 64x32 B tile via GLOBAL_LOAD_ASYNC_TO_LDS_B128 (ASYNCcnt), waiting
// only on the previous chunk while the next one is in flight.
// ---------------------------------------------------------------------------
__global__ __launch_bounds__(256)
void gemm_wmma_kernel(const _Float16* __restrict__ A,
                      const _Float16* __restrict__ Bt,
                      float* __restrict__ C,
                      const float* __restrict__ bias,
                      int M, int N, int K)
{
    const int tid  = threadIdx.x;
    const int wave = tid >> 5, lane = tid & 31;
    const int lane16 = lane & 15, grp = lane >> 4;
    const int nBlocks = N >> 6;
    const int mB = blockIdx.x / nBlocks;   // 64-row block
    const int nB = blockIdx.x % nBlocks;   // 64-col block
    (void)M;

    __shared__ __align__(64) _Float16 aT[2][64][32];
    __shared__ __align__(64) _Float16 bT[2][64][32];

    // staging assignment: thread -> (row, 16B segment)
    const int srow = tid >> 2;           // 0..63
    const int sseg = (tid & 3) * 8;      // halves: 0,8,16,24
    const _Float16* gA = A  + (size_t)(mB * 64 + srow) * K + sseg;
    const _Float16* gB = Bt + (size_t)(nB * 64 + srow) * K + sseg;

    // wave -> two C tiles sharing the same M sub-tile
    const int t0 = wave * 2;
    const int iT = t0 >> 2, jT0 = t0 & 3, jT1 = jT0 + 1;
    v8f acc0 = {}, acc1 = {};

    const int nChunks = K >> 5;
    int cur = 0;

    // prologue: stage chunk 0
    copy16_g2l(&aT[0][srow][sseg], gA);
    copy16_g2l(&bT[0][srow][sseg], gB);

    for (int c = 0; c < nChunks - 1; ++c) {
        const int k1 = (c + 1) << 5;
        copy16_g2l(&aT[cur ^ 1][srow][sseg], gA + k1);
        copy16_g2l(&bT[cur ^ 1][srow][sseg], gB + k1);
        wait_async<2>();          // only the 2 prefetches may remain in flight
        __syncthreads();

        const _Float16* pa = &aT[cur][iT * 16 + lane16][0];
        v8h x = *(const v8h*)(pa + grp * 16);
        v8h y = *(const v8h*)(pa + grp * 16 + 8);
        v16h a = __builtin_shufflevector(x, y, 0,1,2,3,4,5,6,7,8,9,10,11,12,13,14,15);
        v16h bf0 = *(const v16h*)(&bT[cur][jT0 * 16 + lane16][grp * 16]);
        v16h bf1 = *(const v16h*)(&bT[cur][jT1 * 16 + lane16][grp * 16]);
        acc0 = wmma_f16(a, bf0, acc0);
        acc1 = wmma_f16(a, bf1, acc1);
        __syncthreads();          // protect buffer reuse
        cur ^= 1;
    }

    // epilogue chunk
    wait_async<0>();
    __syncthreads();
    {
        const _Float16* pa = &aT[cur][iT * 16 + lane16][0];
        v8h x = *(const v8h*)(pa + grp * 16);
        v8h y = *(const v8h*)(pa + grp * 16 + 8);
        v16h a = __builtin_shufflevector(x, y, 0,1,2,3,4,5,6,7,8,9,10,11,12,13,14,15);
        v16h bf0 = *(const v16h*)(&bT[cur][jT0 * 16 + lane16][grp * 16]);
        v16h bf1 = *(const v16h*)(&bT[cur][jT1 * 16 + lane16][grp * 16]);
        acc0 = wmma_f16(a, bf0, acc0);
        acc1 = wmma_f16(a, bf1, acc1);
    }

    const int n0 = nB * 64 + jT0 * 16 + lane16;
    const int n1 = nB * 64 + jT1 * 16 + lane16;
    const float bv0 = bias[n0], bv1 = bias[n1];
    #pragma unroll
    for (int r = 0; r < 8; ++r) {
        int m = mB * 64 + iT * 16 + grp * 8 + r;
        C[(size_t)m * N + n0] = acc0[r] + bv0;
        C[(size_t)m * N + n1] = acc1[r] + bv1;
    }
}

// ---------------------------------------------------------------------------
// Kernel 3: LayerNorm + ReLU, emitting f16 activations. One block per row.
// ---------------------------------------------------------------------------
__global__ __launch_bounds__(256)
void ln_relu_kernel(const float* __restrict__ X,
                    const float* __restrict__ g,
                    const float* __restrict__ be,
                    _Float16* __restrict__ out, int N)
{
    __shared__ float sbuf[256];
    const int row = blockIdx.x, tid = threadIdx.x;
    const float* x = X + (size_t)row * N;

    float ls = 0.f;
    for (int c = tid; c < N; c += 256) ls += x[c];
    float mu = blockReduce(ls, sbuf, 2) / (float)N;

    float lv = 0.f;
    for (int c = tid; c < N; c += 256) { float d = x[c] - mu; lv += d * d; }
    float var = blockReduce(lv, sbuf, 2) / (float)N;
    float inv = rsqrtf(var + 1e-5f);

    for (int c = tid; c < N; c += 256) {
        float v = (x[c] - mu) * inv * g[c] + be[c];
        out[(size_t)row * N + c] = (_Float16)fmaxf(v, 0.f);
    }
}

// ---------------------------------------------------------------------------
// Kernel 4: f32[K][N] -> f16[N][K] transpose-convert (weights, once per launch)
// ---------------------------------------------------------------------------
__global__ __launch_bounds__(256)
void transpose_f32_f16(const float* __restrict__ in, _Float16* __restrict__ out,
                       int K, int N)
{
    size_t idx = (size_t)blockIdx.x * 256 + threadIdx.x;
    size_t total = (size_t)K * N;
    if (idx < total) {
        int k = (int)(idx / (size_t)N);
        int n = (int)(idx % (size_t)N);
        out[(size_t)n * K + k] = (_Float16)in[idx];
    }
}

extern "C" void kernel_launch(void* const* d_in, const int* in_sizes, int n_in,
                              void* d_out, int out_size, void* d_ws, size_t ws_size,
                              hipStream_t stream) {
    (void)in_sizes; (void)n_in; (void)out_size; (void)ws_size;
    const float* pairs0 = (const float*)d_in[0];   // [256,1024,2]
    const float* pairs1 = (const float*)d_in[1];   // [256,512,2]
    const float* W1 = (const float*)d_in[2];       // [8192,512]
    const float* b1 = (const float*)d_in[3];
    const float* g1 = (const float*)d_in[4];
    const float* be1 = (const float*)d_in[5];
    const float* W2 = (const float*)d_in[6];       // [512,256]
    const float* b2 = (const float*)d_in[7];
    const float* g2 = (const float*)d_in[8];
    const float* be2 = (const float*)d_in[9];
    const float* W3 = (const float*)d_in[10];      // [256,128]
    const float* b3 = (const float*)d_in[11];
    float* out = (float*)d_out;                    // [256,128]

    char* ws = (char*)d_ws;
    _Float16* W1t   = (_Float16*)(ws + 0);          // 512*8192*2  = 8388608
    _Float16* W2t   = (_Float16*)(ws + 8388608);    // 256*512*2   = 262144
    _Float16* W3t   = (_Float16*)(ws + 8650752);    // 128*256*2   = 65536
    _Float16* feats = (_Float16*)(ws + 8716288);    // 256*8192*2  = 4194304
    float*    y1    = (float*)   (ws + 12910592);   // 256*512*4   = 524288
    _Float16* h1    = (_Float16*)(ws + 13434880);   // 256*512*2   = 262144
    float*    y2    = (float*)   (ws + 13697024);   // 256*256*4   = 262144
    _Float16* h2    = (_Float16*)(ws + 13959168);   // 256*256*2   = 131072

    transpose_f32_f16<<<16384, 256, 0, stream>>>(W1, W1t, 8192, 512);
    transpose_f32_f16<<<512,   256, 0, stream>>>(W2, W2t, 512, 256);
    transpose_f32_f16<<<128,   256, 0, stream>>>(W3, W3t, 256, 128);

    pers_image_kernel<<<512, 256, 0, stream>>>(pairs0, pairs1, feats);

    // grids: (M/64) * (N/64) blocks
    gemm_wmma_kernel<<<32, 256, 0, stream>>>(feats, W1t, y1, b1, 256, 512, 8192);
    ln_relu_kernel<<<256, 256, 0, stream>>>(y1, g1, be1, h1, 512);
    gemm_wmma_kernel<<<16, 256, 0, stream>>>(h1, W2t, y2, b2, 256, 256, 512);
    ln_relu_kernel<<<256, 256, 0, stream>>>(y2, g2, be2, h2, 256);
    gemm_wmma_kernel<<<8, 256, 0, stream>>>(h2, W3t, out, b3, 256, 128, 256);
}
